// MCGAT_7619271983396
// MI455X (gfx1250) — matrix operand order repaired and checked by reading, not compiled, not run on hardware
//
#include <hip/hip_runtime.h>
#include <hip/hip_bf16.h>

// ---------------------------------------------------------------------------
// MCGAT fused pipeline for MI455X (gfx1250, wave32, WMMA bf16)
//   BS=64, C=512, D_IN=256, D_OUT=256, H=4, HD=64
// v2: pre-convert h/W to bf16 (removes ~40 v_cvt per 4 wmma from the GEMM hot
// loop, halves its load bytes); stage A_fc tile via global_load_async_to_lds
// (ASYNCcnt path) in the attention kernel.
// ---------------------------------------------------------------------------

#define BSZ   64
#define CCH   512
#define DIN   256
#define DOUT  256
#define NH    4
#define HDIM  64

typedef __attribute__((ext_vector_type(16))) __bf16 v16bf_t;
typedef __attribute__((ext_vector_type(8)))  float  v8f_t;

__device__ __forceinline__ float lrelu02(float x) {
    return x > 0.0f ? x : 0.2f * x;
}

// ---------------------------------------------------------------------------
// Kernel 0: f32 -> bf16 bulk convert (bandwidth bound, L2 resident)
// ---------------------------------------------------------------------------
__global__ __launch_bounds__(256) void k_cvt_bf16(
    const float* __restrict__ src, __bf16* __restrict__ dst, int n4)
{
    const int i = blockIdx.x * 256 + threadIdx.x;
    if (i < n4) {
        const float4 v = ((const float4*)src)[i];
        alignas(8) __bf16 t[4] = {(__bf16)v.x, (__bf16)v.y, (__bf16)v.z, (__bf16)v.w};
        ((uint2*)dst)[i] = *(const uint2*)t;
    }
}

// ---------------------------------------------------------------------------
// Kernel 1: Wh = h @ W^T   (M=BS*C=32768, K=256, N=256), bf16 WMMA.
// Inputs already bf16. Output written transposed: WhT[n][h][d][c] (bf16).
// Block: 256 threads (8 waves). Wave w computes rows [bx*128+w*16, +16),
// cols [by*64, +64) as 4 accumulator tiles.
// ---------------------------------------------------------------------------
__global__ __launch_bounds__(256) void k_gemm_wh(
    const __bf16* __restrict__ hbf, const __bf16* __restrict__ Wbf,
    __bf16* __restrict__ WhT)
{
    const int wave = threadIdx.x >> 5;
    const int lane = threadIdx.x & 31;
    const int half = lane >> 4;
    const int l16  = lane & 15;

    const int m0 = blockIdx.x * 128 + wave * 16;  // flattened (n*C + c) row
    const int n0 = blockIdx.y * 64;               // output-feature col

    v8f_t acc[4] = {};

    for (int k0 = 0; k0 < DIN; k0 += 32) {
        // --- A fragment: row r=l16; K-slots {16h..16h+15} ---
        union { v16bf_t v; uint4 q[2]; } af;
        const __bf16* arow = hbf + (size_t)(m0 + l16) * DIN + k0;
        af.q[0] = *(const uint4*)(arow + 16 * half);       // 8 bf16 = 16B
        af.q[1] = *(const uint4*)(arow + 16 * half + 8);   // 8 bf16 = 16B
        #pragma unroll
        for (int t = 0; t < 4; ++t) {
            // --- B fragment: col nn = n0+t*16+l16; K = 16*half + (0..15) ---
            // B[k][nn] = W^T[k][nn] = W[nn][k] ; 16 contiguous bf16 = 32B
            const __bf16* brow =
                Wbf + (size_t)(n0 + t * 16 + l16) * DIN + k0 + 16 * half;
            v16bf_t bfrag = *(const v16bf_t*)brow;
            acc[t] = __builtin_amdgcn_wmma_f32_16x16x32_bf16(
                false, af.v, false, bfrag, (short)0, acc[t], false, false);
        }
    }

    // --- store transposed bf16: lane's 8 values are consecutive in c ---
    const int mbase = m0 + 8 * half;          // rows mbase..mbase+7 (v=0..7)
    const int n = mbase / CCH;
    const int c = mbase % CCH;                // multiple of 8 -> 16B aligned
    #pragma unroll
    for (int t = 0; t < 4; ++t) {
        const int dout = n0 + t * 16 + l16;
        const int hh = dout >> 6;             // head
        const int d  = dout & 63;             // dim within head
        alignas(16) __bf16 tmp[8];
        #pragma unroll
        for (int v = 0; v < 8; ++v) tmp[v] = (__bf16)acc[t][v];
        *(uint4*)(WhT + ((size_t)((n * NH + hh) * HDIM + d) * CCH + c)) =
            *(const uint4*)tmp;
    }
}

// ---------------------------------------------------------------------------
// Kernel 2: e_i[n,h,c] = sum_d WhT[n,h,d,c]*a[h,d] ; e_j with a[h,HD+d]
// ---------------------------------------------------------------------------
__global__ __launch_bounds__(256) void k_eproj(
    const __bf16* __restrict__ WhT, const float* __restrict__ a,
    float* __restrict__ ei, float* __restrict__ ej)
{
    const int nh = blockIdx.x;                // n*H + h
    const int hh = nh & (NH - 1);
    for (int c = threadIdx.x; c < CCH; c += 256) {
        float si = 0.0f, sj = 0.0f;
        #pragma unroll 4
        for (int d = 0; d < HDIM; ++d) {
            float w = (float)WhT[((size_t)nh * HDIM + d) * CCH + c];
            si = fmaf(w, a[hh * 2 * HDIM + d], si);
            sj = fmaf(w, a[hh * 2 * HDIM + HDIM + d], sj);
        }
        ei[(size_t)nh * CCH + c] = si;
        ej[(size_t)nh * CCH + c] = sj;
    }
}

// ---------------------------------------------------------------------------
// Kernel 3: fused attention.  Block = (n, i-tile of 16 rows), 128 thr = 4
// waves, wave w == head w.  A_fc row-tile staged once into LDS via
// GLOBAL_LOAD_ASYNC_TO_LDS_B128 (ASYNCcnt), shared by all 4 heads; padded
// stride 516 kills the 16-way bank conflict of the 512 stride.
// Two-pass softmax; aggregation attn @ Wh via bf16 WMMA, P fragment built
// directly in A-matrix register layout.
// ---------------------------------------------------------------------------
__global__ __launch_bounds__(128) void k_attn(
    const float* __restrict__ A_fc, const __bf16* __restrict__ WhT,
    const float* __restrict__ ei, const float* __restrict__ ej,
    float* __restrict__ h_prime)
{
    __shared__ float sAfc[16][516];    // 33.0 KB (padded)
    __shared__ float sEj[NH][CCH];     //  8.0 KB
    __shared__ float sEi[NH][16];
    __shared__ float sRS[NH][16];      // per-row softmax denominators

    const int n  = blockIdx.x;
    const int i0 = blockIdx.y * 16;
    const int hh   = threadIdx.x >> 5;   // wave == head
    const int lane = threadIdx.x & 31;
    const int half = lane >> 4;
    const int l16  = lane & 15;

    // ---- stage A_fc tile: async global -> LDS, 16 x (128 lanes x 16B) ----
    // Global rows are contiguous (row stride == C), so source is linear.
    const float* Afc_src = A_fc + ((size_t)n * CCH + i0) * CCH;  // uniform
    #pragma unroll 4
    for (int it = 0; it < 16; ++it) {
        const unsigned e4 = it * 128 + threadIdx.x;   // float4 index
        const unsigned r  = e4 >> 7;                  // row (128 float4 / row)
        const unsigned j  = (e4 & 127) * 4;           // col
        const unsigned lds  = (unsigned)(uintptr_t)&sAfc[r][j];
        const unsigned voff = e4 * 16;                // global byte offset
        asm volatile("global_load_async_to_lds_b128 %0, %1, %2 offset:0"
                     :: "v"(lds), "v"(voff), "s"(Afc_src) : "memory");
    }
    // ---- e_j / e_i via normal loads while the async copies fly ----
    for (int idx = threadIdx.x; idx < NH * CCH; idx += 128) {
        sEj[idx >> 9][idx & 511] =
            ej[((size_t)n * NH + (idx >> 9)) * CCH + (idx & 511)];
    }
    if (threadIdx.x < NH * 16) {
        const int h2 = threadIdx.x >> 4, r = threadIdx.x & 15;
        sEi[h2][r] = ei[((size_t)n * NH + h2) * CCH + i0 + r];
    }
    asm volatile("s_wait_asynccnt 0" ::: "memory");
    __syncthreads();

    const float ei_r = sEi[hh][l16];     // my row's e_i

    // ---- pass 1: row max (lane pair splits j range, combine via shfl) ----
    float m = -3.0e38f;
    for (int j = half * 256; j < half * 256 + 256; ++j) {
        float s = lrelu02(ei_r + sEj[hh][j]) * sAfc[l16][j];
        m = fmaxf(m, s);
    }
    m = fmaxf(m, __shfl_xor(m, 16, 32));

    // ---- pass 2: exp + WMMA accumulate over j tiles of 32 ----
    v8f_t acc[4] = {};
    float rsum = 0.0f;
    const __bf16* wbase = WhT + (size_t)(n * NH + hh) * HDIM * CCH;

    for (int jt = 0; jt < CCH; jt += 32) {
        // P fragment, built directly in 16x32 bf16 A-matrix lane layout:
        //   elems 0..15 -> K = 16*half + i
        v16bf_t p;
        #pragma unroll
        for (int i = 0; i < 16; ++i) {
            const int k = 16 * half + i;
            const int j = jt + k;
            const float s = lrelu02(ei_r + sEj[hh][j]) * sAfc[l16][j];
            const float e = __expf(s - m);
            rsum += e;
            p[i] = (__bf16)e;
        }
        // B fragments: WhT[n][hh][d][j], K=j contiguous -> 32B vector loads
        #pragma unroll
        for (int t = 0; t < 4; ++t) {
            const __bf16* src = wbase + (size_t)(t * 16 + l16) * CCH + jt + 16 * half;
            v16bf_t b = *(const v16bf_t*)src;
            acc[t] = __builtin_amdgcn_wmma_f32_16x16x32_bf16(
                false, p, false, b, (short)0, acc[t], false, false);
        }
    }

    // ---- softmax denominators: combine lane pairs, broadcast via LDS ----
    rsum += __shfl_xor(rsum, 16, 32);
    if (half == 0) sRS[hh][l16] = rsum;
    __syncthreads();

    // ---- normalize + store h_prime[n][i][hh*64 + d] (f32) ----
    float* hp = h_prime + ((size_t)n * CCH + i0) * DOUT + hh * HDIM;
    #pragma unroll
    for (int t = 0; t < 4; ++t) {
        #pragma unroll
        for (int v = 0; v < 8; ++v) {
            const int r2 = v + 8 * half;                 // row in tile
            const float val = acc[t][v] / sRS[hh][r2];
            hp[(size_t)r2 * DOUT + t * 16 + l16] = val;
        }
    }
}

// ---------------------------------------------------------------------------
// Kernel 4: channel-gating MLP + softmax over channels + weighted sum.
// ---------------------------------------------------------------------------
__global__ __launch_bounds__(256) void k_gate(
    const float* __restrict__ h_prime, const float* __restrict__ W1,
    const float* __restrict__ b1, const float* __restrict__ W2,
    const float* __restrict__ b2, float* __restrict__ out)
{
    __shared__ float sS[CCH];      // scores, then softmax weights
    __shared__ float sRed[256];

    const int n = blockIdx.x;
    const int wave = threadIdx.x >> 5;
    const int lane = threadIdx.x & 31;
    const float* hpn = h_prime + (size_t)n * CCH * DOUT;

    // phase 1: s[c] = W2 @ tanh(W1 @ h_prime[n,c] + b1) + b2
    for (int c = wave; c < CCH; c += 8) {
        const float* hc = hpn + (size_t)c * DOUT;
        float partial = 0.0f;
        #pragma unroll
        for (int uu = 0; uu < 2; ++uu) {
            const int u = lane + uu * 32;
            float acc = b1[u];
            #pragma unroll 4
            for (int d = 0; d < DOUT; ++d)
                acc = fmaf(hc[d], W1[u * DOUT + d], acc);
            partial += tanhf(acc) * W2[u];
        }
        #pragma unroll
        for (int off = 16; off > 0; off >>= 1)
            partial += __shfl_xor(partial, off, 32);
        if (lane == 0) sS[c] = partial + b2[0];
    }
    __syncthreads();

    // phase 2: softmax over the 512 channels
    const float s0 = sS[threadIdx.x];
    const float s1 = sS[threadIdx.x + 256];
    sRed[threadIdx.x] = fmaxf(s0, s1);
    __syncthreads();
    for (int off = 128; off > 0; off >>= 1) {
        if (threadIdx.x < off)
            sRed[threadIdx.x] = fmaxf(sRed[threadIdx.x], sRed[threadIdx.x + off]);
        __syncthreads();
    }
    const float m = sRed[0];
    __syncthreads();
    const float e0 = __expf(s0 - m), e1 = __expf(s1 - m);
    sRed[threadIdx.x] = e0 + e1;
    __syncthreads();
    for (int off = 128; off > 0; off >>= 1) {
        if (threadIdx.x < off)
            sRed[threadIdx.x] += sRed[threadIdx.x + off];
        __syncthreads();
    }
    const float inv = 1.0f / sRed[0];
    __syncthreads();
    sS[threadIdx.x]       = e0 * inv;
    sS[threadIdx.x + 256] = e1 * inv;
    __syncthreads();

    // phase 3: out[n,d] = sum_c w[c] * h_prime[n,c,d]
    const int d = threadIdx.x;
    float acc = 0.0f;
    for (int c = 0; c < CCH; ++c)
        acc = fmaf(sS[c], hpn[(size_t)c * DOUT + d], acc);
    out[(size_t)n * DOUT + d] = acc;
}

// ---------------------------------------------------------------------------
// Launch
// ---------------------------------------------------------------------------
extern "C" void kernel_launch(void* const* d_in, const int* in_sizes, int n_in,
                              void* d_out, int out_size, void* d_ws, size_t ws_size,
                              hipStream_t stream) {
    const float* h    = (const float*)d_in[0];
    const float* A_fc = (const float*)d_in[1];
    const float* W    = (const float*)d_in[2];
    const float* a    = (const float*)d_in[3];
    const float* W1   = (const float*)d_in[4];
    const float* b1   = (const float*)d_in[5];
    const float* W2   = (const float*)d_in[6];
    const float* b2   = (const float*)d_in[7];
    float* out = (float*)d_out;

    // workspace layout
    const size_t WHT_BYTES = (size_t)BSZ * NH * HDIM * CCH * sizeof(__bf16); // 16.8 MB
    const size_t HP_BYTES  = (size_t)BSZ * CCH * DOUT * sizeof(float);       // 33.6 MB
    const size_t E_BYTES   = (size_t)BSZ * NH * CCH * sizeof(float);         // 0.5 MB
    const size_t HBF_BYTES = (size_t)BSZ * CCH * DIN * sizeof(__bf16);       // 16.8 MB
    char* ws = (char*)d_ws;
    size_t off = 0;
    __bf16* WhT    = (__bf16*)(ws + off); off += WHT_BYTES;
    float*  hprime = (float*)(ws + off);  off += HP_BYTES;
    float*  ei     = (float*)(ws + off);  off += E_BYTES;
    float*  ej     = (float*)(ws + off);  off += E_BYTES;
    __bf16* hbf    = (__bf16*)(ws + off); off += HBF_BYTES;
    __bf16* Wbf    = (__bf16*)(ws + off);

    const int h_n4 = BSZ * CCH * DIN / 4;   // 2,097,152 float4
    const int w_n4 = DOUT * DIN / 4;        // 16,384 float4
    k_cvt_bf16<<<dim3(h_n4 / 256), 256, 0, stream>>>(h, hbf, h_n4);
    k_cvt_bf16<<<dim3(w_n4 / 256), 256, 0, stream>>>(W, Wbf, w_n4);

    k_gemm_wh<<<dim3((BSZ * CCH) / 128, DOUT / 64), 256, 0, stream>>>(hbf, Wbf, WhT);
    k_eproj<<<dim3(BSZ * NH), 256, 0, stream>>>(WhT, a, ei, ej);
    k_attn<<<dim3(BSZ, CCH / 16), 128, 0, stream>>>(A_fc, WhT, ei, ej, hprime);
    k_gate<<<dim3(BSZ), 256, 0, stream>>>(hprime, W1, b1, W2, b2, out);
}
